// SelfAttention_5488968204451
// MI455X (gfx1250) — matrix operand order repaired and checked by reading, not compiled, not run on hardware
//
#include <hip/hip_runtime.h>

typedef __attribute__((ext_vector_type(16))) _Float16 v16h;
typedef __attribute__((ext_vector_type(8)))  _Float16 v8h;
typedef __attribute__((ext_vector_type(4)))  _Float16 v4h;
typedef __attribute__((ext_vector_type(8)))  float    v8f;

#define BATCH 2
#define SEQ   2048
#define HEADS 16
#define HDIM  64
#define EMBED 1024

// ---------------------------------------------------------------------------
// WMMA fragment gathers per CDNA5 ISA layouts (CDNA5 ISA §7.12.2).
// A 16x32 f16: lane row = lane&15; half j -> K = kbase + (lane>=16?8:0) + (j<8 ? j : j+8)
// B 32x16 f16: lane col = lane&15; half j -> K = kbase + (lane>=16?8:0) + (j<8 ? j : j+8)
// C 16x16 f32: v8f element i -> (m = i + (lane>=16?8:0), n = lane&15)
// Both gathers read contiguous 16B/32B runs -> lower to ds_load_b128.
// ---------------------------------------------------------------------------
__device__ __forceinline__ v16h gather_a16(const _Float16* src, int ld, int lane, int kbase) {
  const int row  = lane & 15;
  const int koff = ((lane >> 4) & 1) * 8;
  v16h a;
#pragma unroll
  for (int j = 0; j < 16; ++j) {
    int k = kbase + koff + ((j < 8) ? j : (j + 8));
    a[j] = src[row * ld + k];
  }
  return a;
}

// B(k, n) stored as src[n * ld + k] (contiguous in k)
__device__ __forceinline__ v16h gather_b_nk(const _Float16* src, int ld, int lane,
                                            int kbase, int nbase) {
  const int n    = nbase + (lane & 15);
  const int koff = kbase + ((lane >> 4) & 1) * 8;
  v16h b;
#pragma unroll
  for (int j = 0; j < 16; ++j) b[j] = src[n * ld + koff + ((j < 8) ? j : (j + 8))];
  return b;
}

__device__ __forceinline__ v8f wmma_f16(v16h a, v16h b, v8f c) {
  return __builtin_amdgcn_wmma_f32_16x16x32_f16(false, a, false, b, (short)0, c, false, false);
}

// ---------------------------------------------------------------------------
// CDNA5 async global->LDS copy (ASYNCcnt tracked), CDNA5 ISA.
// Generic pointer to LDS: low 32 bits are the LDS offset (aperture mapping).
// ---------------------------------------------------------------------------
__device__ __forceinline__ uint32_t lds32(const void* p) {
  return (uint32_t)(uintptr_t)p;
}
__device__ __forceinline__ void async_b128(uint32_t lds, const void* g) {
  asm volatile("global_load_async_to_lds_b128 %0, %1, off"
               :: "v"(lds), "v"(g) : "memory");
}
#define WAIT_ASYNC_LE(n) asm volatile("s_wait_asynccnt " #n ::: "memory")

// ---------------------------------------------------------------------------
// Kernel 0: fp32 -> f16 bulk convert (weights), float4 -> v4h per thread.
// ---------------------------------------------------------------------------
__global__ __launch_bounds__(256) void k_cvt_f16(const float* __restrict__ src,
                                                 _Float16* __restrict__ dst, int n4) {
  int i = blockIdx.x * 256 + threadIdx.x;
  if (i < n4) {
    float4 f = ((const float4*)src)[i];
    v4h h = {(_Float16)f.x, (_Float16)f.y, (_Float16)f.z, (_Float16)f.w};
    ((v4h*)dst)[i] = h;
  }
}

// ---------------------------------------------------------------------------
// Kernel 1: per-head projection  Y[...] = X @ W^T (shared 64x64 weight).
// TRANSPOSE=0: Y[b,h][s][d]  (Q, K).  TRANSPOSE=1: Y[b,h][d][s]  (V).
// ---------------------------------------------------------------------------
template <bool TRANSPOSE>
__global__ __launch_bounds__(32) void k_proj(const float* __restrict__ X,
                                             const _Float16* __restrict__ Wh,
                                             _Float16* __restrict__ Y) {
  __shared__ __align__(16) _Float16 x_s[16 * 64];
  __shared__ __align__(16) _Float16 w_s[64 * 64];
  const int lane = threadIdx.x;
  const int s0 = blockIdx.x * 16;
  const int h  = blockIdx.y;
  const int b  = blockIdx.z;

#pragma unroll
  for (int i = 0; i < 8; ++i) {            // 16x64 X tile: 256 float4 chunks
    int idx = lane + i * 32;
    int row = idx >> 4, d4 = (idx & 15) * 4;
    float4 f = *(const float4*)(X + ((size_t)(b * SEQ + s0 + row)) * EMBED + h * HDIM + d4);
    v4h hv = {(_Float16)f.x, (_Float16)f.y, (_Float16)f.z, (_Float16)f.w};
    *(v4h*)&x_s[row * 64 + d4] = hv;
  }
#pragma unroll
  for (int i = 0; i < 16; ++i) {           // 64x64 half weight: 512 uint4
    int cc = lane + i * 32;
    ((uint4*)w_s)[cc] = ((const uint4*)Wh)[cc];
  }
  __syncthreads();

  const int n    = lane & 15;
  const int mrow = ((lane >> 4) & 1) * 8;
#pragma unroll
  for (int nt = 0; nt < 4; ++nt) {
    v8f c = {};
#pragma unroll
    for (int ks = 0; ks < 2; ++ks) {
      v16h a  = gather_a16(x_s, 64, lane, ks * 32);
      v16h bm = gather_b_nk(w_s, 64, lane, ks * 32, nt * 16);   // B(d,e)=W[e][d]
      c = wmma_f16(a, bm, c);
    }
    if (TRANSPOSE) {
      _Float16* yb = Y + ((size_t)(b * HEADS + h)) * HDIM * SEQ;
      v8h hv;
#pragma unroll
      for (int i = 0; i < 8; ++i) hv[i] = (_Float16)c[i];
      *(v8h*)&yb[(size_t)(nt * 16 + n) * SEQ + s0 + mrow] = hv;  // contiguous in s
    } else {
      _Float16* yb = Y + (((size_t)(b * HEADS + h)) * SEQ + s0) * HDIM;
#pragma unroll
      for (int i = 0; i < 8; ++i)
        yb[(size_t)(mrow + i) * HDIM + nt * 16 + n] = (_Float16)c[i];
    }
  }
}

// ---------------------------------------------------------------------------
// Kernel 2: flash attention, 32-query tile per wave, online softmax.
// K tiles staged [key][64], V staged transposed [d][32]; both double-buffered
// via global_load_async_to_lds_b128 (16 async ops per tile per lane-set).
// ---------------------------------------------------------------------------
__device__ __forceinline__ void attn_issue_tile(const _Float16* Kh, const _Float16* Vth,
                                                int k0, int lane,
                                                _Float16* k_buf, _Float16* v_buf) {
#pragma unroll
  for (int i = 0; i < 8; ++i) {
    int cc = lane + i * 32;                 // 256 x 16B chunks each
    {
      int key = cc >> 3, part = cc & 7;     // K tile: 32 keys x 64 d
      async_b128(lds32(k_buf + cc * 8),
                 Kh + (size_t)(k0 + key) * HDIM + part * 8);
    }
    {
      int d = cc >> 2, part = cc & 3;       // V tile: 64 d x 32 keys (transposed src)
      async_b128(lds32(v_buf + cc * 8),
                 Vth + (size_t)d * SEQ + k0 + part * 8);
    }
  }
}

__global__ __launch_bounds__(32) void k_attn(const _Float16* __restrict__ Qp,
                                             const _Float16* __restrict__ Kp,
                                             const _Float16* __restrict__ Vt,
                                             const int* __restrict__ mask,
                                             _Float16* __restrict__ AO) {
  __shared__ __align__(16) _Float16 q_s[32 * 64];
  __shared__ __align__(16) _Float16 k_sb[2][32 * 64];
  __shared__ __align__(16) _Float16 v_sb[2][64 * 32];
  __shared__ float    e_s[32 * 32];
  __shared__ _Float16 p_s[32 * 32];
  __shared__ float row_m[32], row_l[32], corr_s[32], e_bias[32];

  const int lane = threadIdx.x;
  const int s0 = blockIdx.x * 32;
  const int h  = blockIdx.y;
  const int b  = blockIdx.z;
  const size_t head_off = ((size_t)(b * HEADS + h)) * SEQ * HDIM;
  const _Float16* Qh  = Qp + head_off + (size_t)s0 * HDIM;
  const _Float16* Kh  = Kp + head_off;
  const _Float16* Vth = Vt + head_off;     // [d][s]

  {
    const uint4* src = (const uint4*)Qh;   // 32x64 halves = 256 uint4
    uint4* dst = (uint4*)q_s;
#pragma unroll
    for (int i = 0; i < 8; ++i) dst[lane + i * 32] = src[lane + i * 32];
  }
  row_m[lane] = -__builtin_inff();
  row_l[lane] = 0.f;
  v8f o[2][4] = {{v8f{}, v8f{}, v8f{}, v8f{}}, {v8f{}, v8f{}, v8f{}, v8f{}}};

  const int n    = lane & 15;
  const int mrow = ((lane >> 4) & 1) * 8;

  attn_issue_tile(Kh, Vth, 0, lane, k_sb[0], v_sb[0]);

  for (int k0 = 0; k0 < SEQ; k0 += 32) {
    const int cur = (k0 >> 5) & 1;
    e_bias[lane] = (mask[b * SEQ + k0 + lane] == 0) ? -1e20f : 0.f;
    if (k0 + 32 < SEQ) {
      attn_issue_tile(Kh, Vth, k0 + 32, lane, k_sb[cur ^ 1], v_sb[cur ^ 1]);
      WAIT_ASYNC_LE(0x10);                 // current tile resident, next in flight
    } else {
      WAIT_ASYNC_LE(0x0);
    }
    __syncthreads();

    // energy 32(q) x 32(key): contraction over d=64
#pragma unroll
    for (int qt = 0; qt < 2; ++qt) {
#pragma unroll
      for (int nt = 0; nt < 2; ++nt) {
        v8f e = {};
#pragma unroll
        for (int ks = 0; ks < 2; ++ks) {
          v16h a  = gather_a16(q_s + qt * 16 * 64, 64, lane, ks * 32);
          v16h bm = gather_b_nk(k_sb[cur], 64, lane, ks * 32, nt * 16);
          e = wmma_f16(a, bm, e);
        }
#pragma unroll
        for (int i = 0; i < 8; ++i)
          e_s[(qt * 16 + mrow + i) * 32 + nt * 16 + n] = e[i];
      }
    }
    __syncthreads();

    // online softmax: lane r handles query row r (all 32 lanes active)
    {
      const int r = lane;
      float mold = row_m[r];
      float tmax = -__builtin_inff();
      float ev[32];
#pragma unroll
      for (int c = 0; c < 32; ++c) {
        float val = (e_s[r * 32 + c] + e_bias[c]) * 0.03125f;  // mask, then 1/sqrt(1024)
        ev[c] = val;
        tmax = fmaxf(tmax, val);
      }
      float mnew = fmaxf(mold, tmax);
      float corr = __expf(mold - mnew);
      float lsum = 0.f;
#pragma unroll
      for (int c = 0; c < 32; ++c) {
        float p = __expf(ev[c] - mnew);
        lsum += p;
        p_s[r * 32 + c] = (_Float16)p;
      }
      row_m[r] = mnew;
      row_l[r] = row_l[r] * corr + lsum;
      corr_s[r] = corr;
    }
    __syncthreads();

    // rescale accumulators, then O += P(32x32) * V(32x64)
#pragma unroll
    for (int qt = 0; qt < 2; ++qt) {
      float cr[8];
#pragma unroll
      for (int i = 0; i < 8; ++i) cr[i] = corr_s[qt * 16 + mrow + i];
#pragma unroll
      for (int nt = 0; nt < 4; ++nt)
#pragma unroll
        for (int i = 0; i < 8; ++i) o[qt][nt][i] *= cr[i];

      v16h a = gather_a16(p_s + qt * 16 * 32, 32, lane, 0);
#pragma unroll
      for (int nt = 0; nt < 4; ++nt) {
        v16h bm = gather_b_nk(v_sb[cur], 32, lane, 0, nt * 16); // B(key,d)=Vt[d][key]
        o[qt][nt] = wmma_f16(a, bm, o[qt][nt]);
      }
    }
    __syncthreads();
  }

  // normalize, store f16 in [B, S, EMBED]
#pragma unroll
  for (int qt = 0; qt < 2; ++qt) {
    float invl[8];
#pragma unroll
    for (int i = 0; i < 8; ++i) invl[i] = 1.f / row_l[qt * 16 + mrow + i];
#pragma unroll
    for (int nt = 0; nt < 4; ++nt)
#pragma unroll
      for (int i = 0; i < 8; ++i)
        AO[((size_t)(b * SEQ + s0 + qt * 16 + mrow + i)) * EMBED + h * HDIM + nt * 16 + n] =
            (_Float16)(o[qt][nt][i] * invl[i]);
  }
}

// ---------------------------------------------------------------------------
// Kernel 3: Out = AO(4096x1024) @ WoH^T + bo (fp32 out). 32x64 tile per wave,
// async double-buffered A/B staging (12 async ops per k-tile).
// ---------------------------------------------------------------------------
__device__ __forceinline__ void outproj_issue(const _Float16* AO_r, const _Float16* Wt_r,
                                              int k0, int lane,
                                              _Float16* a_b, _Float16* b_b) {
#pragma unroll
  for (int i = 0; i < 4; ++i) {            // A: 32 rows x 32 halves = 128 chunks
    int cc = lane + i * 32;
    int row = cc >> 2, part = cc & 3;
    async_b128(lds32(a_b + cc * 8), AO_r + (size_t)row * EMBED + k0 + part * 8);
  }
#pragma unroll
  for (int i = 0; i < 8; ++i) {            // B: 64 rows x 32 halves = 256 chunks
    int cc = lane + i * 32;
    int nn = cc >> 2, part = cc & 3;
    async_b128(lds32(b_b + cc * 8), Wt_r + (size_t)nn * EMBED + k0 + part * 8);
  }
}

__global__ __launch_bounds__(32) void k_outproj(const _Float16* __restrict__ AO,
                                                const _Float16* __restrict__ WoH,
                                                const float* __restrict__ bo,
                                                float* __restrict__ Out) {
  __shared__ __align__(16) _Float16 a_sb[2][32 * 32];
  __shared__ __align__(16) _Float16 b_sb[2][64 * 32];
  const int lane = threadIdx.x;
  const int m0 = blockIdx.x * 32;          // row in flattened [B*S]
  const int n0 = blockIdx.y * 64;
  const _Float16* AO_r = AO + (size_t)m0 * EMBED;
  const _Float16* Wt_r = WoH + (size_t)n0 * EMBED;
  v8f acc[2][4] = {{v8f{}, v8f{}, v8f{}, v8f{}}, {v8f{}, v8f{}, v8f{}, v8f{}}};

  outproj_issue(AO_r, Wt_r, 0, lane, a_sb[0], b_sb[0]);

  for (int k0 = 0; k0 < EMBED; k0 += 32) {
    const int cur = (k0 >> 5) & 1;
    if (k0 + 32 < EMBED) {
      outproj_issue(AO_r, Wt_r, k0 + 32, lane, a_sb[cur ^ 1], b_sb[cur ^ 1]);
      WAIT_ASYNC_LE(0xC);
    } else {
      WAIT_ASYNC_LE(0x0);
    }
    __syncthreads();
#pragma unroll
    for (int mt = 0; mt < 2; ++mt) {
      v16h a = gather_a16(a_sb[cur] + mt * 16 * 32, 32, lane, 0);
#pragma unroll
      for (int nt = 0; nt < 4; ++nt) {
        v16h bm = gather_b_nk(b_sb[cur], 32, lane, 0, nt * 16);
        acc[mt][nt] = wmma_f16(a, bm, acc[mt][nt]);
      }
    }
    __syncthreads();
  }

  const int n    = lane & 15;
  const int mrow = ((lane >> 4) & 1) * 8;
#pragma unroll
  for (int mt = 0; mt < 2; ++mt)
#pragma unroll
    for (int nt = 0; nt < 4; ++nt) {
      const int col = n0 + nt * 16 + n;
      const float bias = bo[col];
#pragma unroll
      for (int i = 0; i < 8; ++i)
        Out[((size_t)(m0 + mt * 16 + mrow + i)) * EMBED + col] = acc[mt][nt][i] + bias;
    }
}

// ---------------------------------------------------------------------------
extern "C" void kernel_launch(void* const* d_in, const int* in_sizes, int n_in,
                              void* d_out, int out_size, void* d_ws, size_t ws_size,
                              hipStream_t stream) {
  const float* values = (const float*)d_in[0];
  const float* keys   = (const float*)d_in[1];
  const float* query  = (const float*)d_in[2];
  const int*   mask   = (const int*)d_in[3];
  const float* Wv     = (const float*)d_in[4];
  const float* Wk     = (const float*)d_in[5];
  const float* Wq     = (const float*)d_in[6];
  const float* Wo     = (const float*)d_in[7];
  const float* bo     = (const float*)d_in[8];
  float* out = (float*)d_out;

  const size_t HE = (size_t)BATCH * HEADS * SEQ * HDIM;    // 4,194,304 halves
  _Float16* ws  = (_Float16*)d_ws;
  _Float16* Vt  = ws;                 // [b,h][d][s]
  _Float16* Kp  = ws + HE;            // [b,h][s][d]
  _Float16* Qp  = ws + 2 * HE;        // [b,h][s][d]
  _Float16* AO  = ws + 3 * HE;        // [b*s][e]
  _Float16* WoH = ws + 4 * HE;        // 1M halves
  _Float16* WvH = WoH + (size_t)EMBED * EMBED;
  _Float16* WkH = WvH + HDIM * HDIM;
  _Float16* WqH = WkH + HDIM * HDIM;

  dim3 blk(32, 1, 1);
  // weight conversions
  k_cvt_f16<<<dim3((EMBED * EMBED / 4 + 255) / 256), dim3(256), 0, stream>>>(Wo, WoH, EMBED * EMBED / 4);
  k_cvt_f16<<<dim3(4), dim3(256), 0, stream>>>(Wv, WvH, HDIM * HDIM / 4);
  k_cvt_f16<<<dim3(4), dim3(256), 0, stream>>>(Wk, WkH, HDIM * HDIM / 4);
  k_cvt_f16<<<dim3(4), dim3(256), 0, stream>>>(Wq, WqH, HDIM * HDIM / 4);

  dim3 gProj(SEQ / 16, HEADS, BATCH);
  k_proj<true ><<<gProj, blk, 0, stream>>>(values, WvH, Vt);
  k_proj<false><<<gProj, blk, 0, stream>>>(keys,   WkH, Kp);
  k_proj<false><<<gProj, blk, 0, stream>>>(query,  WqH, Qp);

  dim3 gAttn(SEQ / 32, HEADS, BATCH);
  k_attn<<<gAttn, blk, 0, stream>>>(Qp, Kp, Vt, mask, AO);

  dim3 gOut((BATCH * SEQ) / 32, EMBED / 64, 1);
  k_outproj<<<gOut, blk, 0, stream>>>(AO, WoH, bo, out);
}
